// Attention_11940009083141
// MI455X (gfx1250) — compile-verified, matrix operand order checked by reading, not run on hardware
//
#include <hip/hip_runtime.h>
#include <hip/hip_bf16.h>

// ---------------------------------------------------------------------------
// Types for WMMA (CDNA5 / gfx1250, wave32)
// ---------------------------------------------------------------------------
typedef __attribute__((ext_vector_type(16))) _Float16 v16h;
typedef __attribute__((ext_vector_type(8)))  _Float16 v8h;
typedef __attribute__((ext_vector_type(8)))  float    v8f;

__device__ __forceinline__ v16h mk16(v8h lo, v8h hi) {
    union { v16h v; v8h h[2]; } u;
    u.h[0] = lo; u.h[1] = hi;
    return u.v;
}

__device__ __forceinline__ v8h cvt8(float4 f0, float4 f1) {
    v8h h;
    h[0] = (_Float16)f0.x; h[1] = (_Float16)f0.y;
    h[2] = (_Float16)f0.z; h[3] = (_Float16)f0.w;
    h[4] = (_Float16)f1.x; h[5] = (_Float16)f1.y;
    h[6] = (_Float16)f1.z; h[7] = (_Float16)f1.w;
    return h;
}

__device__ __forceinline__ v8f wmma_f16(v16h a, v16h b, v8f c) {
    // D = A(16x32 f16) * B(32x16 f16) + C(16x16 f32)
    return __builtin_amdgcn_wmma_f32_16x16x32_f16(
        /*neg_a=*/false, a, /*neg_b=*/false, b,
        /*c_mod=*/(short)0, c, /*reuse_a=*/false, /*reuse_b=*/false);
}

// ---------------------------------------------------------------------------
// Async global->LDS staging (gfx1250 ASYNCcnt path). INST_OFFSET is added to
// BOTH the LDS and global address, so one base covers a contiguous row chunk.
// ---------------------------------------------------------------------------
__device__ __forceinline__ void async_copy32B(unsigned lds, unsigned long long g) {
    asm volatile(
        "global_load_async_to_lds_b128 %0, %1, off\n\t"
        "global_load_async_to_lds_b128 %0, %1, off offset:16"
        :: "v"(lds), "v"(g) : "memory");
}
__device__ __forceinline__ void async_copy64B(unsigned lds, unsigned long long g) {
    asm volatile(
        "global_load_async_to_lds_b128 %0, %1, off\n\t"
        "global_load_async_to_lds_b128 %0, %1, off offset:16\n\t"
        "global_load_async_to_lds_b128 %0, %1, off offset:32\n\t"
        "global_load_async_to_lds_b128 %0, %1, off offset:48"
        :: "v"(lds), "v"(g) : "memory");
}
__device__ __forceinline__ void async_wait0() {
    asm volatile("s_wait_asynccnt 0" ::: "memory");
}

// ---------------------------------------------------------------------------
// Problem constants
// ---------------------------------------------------------------------------
constexpr int BATCH = 2;
constexpr int SEQ   = 2048;
constexpr int HID   = 1024;
constexpr int HEADS = 16;
constexpr int HDIM  = 64;
constexpr int GM = BATCH * SEQ;   // 4096
constexpr int GN = HID;           // 1024
constexpr int GK = HID;           // 1024

// ---------------------------------------------------------------------------
// fp32 -> fp16 bulk conversion (one shot; makes all GEMM operands async-able)
// ---------------------------------------------------------------------------
__global__ __launch_bounds__(256)
void cvt_f16(const float* __restrict__ src, _Float16* __restrict__ dst) {
    const size_t i = (size_t)blockIdx.x * 256 + threadIdx.x;   // one v8 per thread
    const float4* s = (const float4*)src;
    float4 f0 = s[2 * i], f1 = s[2 * i + 1];
    *(v8h*)(dst + 8 * i) = cvt8(f0, f1);
}

// ---------------------------------------------------------------------------
// GEMM: C[M,N] = A[M,K] * W[N,K]^T   (all operands f16, C f32)
// Block tile 128x128x32, 256 threads = 8 waves, wave tile 32x64 (2x4 WMMA).
// LDS rows padded to 40 halves (80B stride -> conflict-free 16B frag reads).
// Double-buffered via async-to-LDS DMA; zero VGPR staging.
// ---------------------------------------------------------------------------
constexpr int BM = 128, BN = 128, BK = 32, LDT = 40;
constexpr int NKT = GK / BK;      // 32

__global__ __launch_bounds__(256)
void gemm_bt(const _Float16* __restrict__ Ah,
             const _Float16* __restrict__ Bh,
             float* __restrict__ C) {
    __shared__ _Float16 As[2][BM * LDT];
    __shared__ _Float16 Bs[2][BM * LDT];

    const int tid = threadIdx.x;
    const int m0 = blockIdx.y * BM;
    const int n0 = blockIdx.x * BN;
    const int lr = tid >> 1;           // 0..127 (tile row)
    const int lc = (tid & 1) * 16;     // 0 or 16 (k sub-chunk)

    const int wave = tid >> 5, lane = tid & 31;
    const int wm = (wave >> 1) * 32;
    const int wn = (wave & 1) * 64;
    const int l16 = lane & 15;
    const int kh8 = (lane >> 4) * 8;

    const _Float16* ga = Ah + (size_t)(m0 + lr) * GK + lc;
    const _Float16* gb = Bh + (size_t)(n0 + lr) * GK + lc;

    auto issue = [&](int kt, int buf) {
        const unsigned la = (unsigned)(size_t)&As[buf][lr * LDT + lc];
        const unsigned lb = (unsigned)(size_t)&Bs[buf][lr * LDT + lc];
        async_copy32B(la, (unsigned long long)(size_t)(ga + kt * BK));
        async_copy32B(lb, (unsigned long long)(size_t)(gb + kt * BK));
    };

    v8f acc[2][4] = {};

    issue(0, 0);
    async_wait0();
    __syncthreads();

    int cur = 0;
    for (int kt = 0; kt < NKT; ++kt) {
        if (kt + 1 < NKT) issue(kt + 1, cur ^ 1);

        v16h af[2], bf[4];
#pragma unroll
        for (int i = 0; i < 2; ++i) {
            const _Float16* base = &As[cur][(wm + 16 * i + l16) * LDT];
            af[i] = mk16(*(const v8h*)(base + kh8), *(const v8h*)(base + 16 + kh8));
        }
#pragma unroll
        for (int j = 0; j < 4; ++j) {
            const _Float16* base = &Bs[cur][(wn + 16 * j + l16) * LDT];
            bf[j] = mk16(*(const v8h*)(base + kh8), *(const v8h*)(base + 16 + kh8));
        }
#pragma unroll
        for (int i = 0; i < 2; ++i)
#pragma unroll
            for (int j = 0; j < 4; ++j)
                acc[i][j] = wmma_f16(af[i], bf[j], acc[i][j]);

        if (kt + 1 < NKT) async_wait0();
        __syncthreads();
        cur ^= 1;
    }

    const int rgrp = (lane >> 4) * 8;
#pragma unroll
    for (int i = 0; i < 2; ++i)
#pragma unroll
        for (int j = 0; j < 4; ++j)
#pragma unroll
            for (int r = 0; r < 8; ++r) {
                const int row = m0 + wm + 16 * i + r + rgrp;
                const int col = n0 + wn + 16 * j + l16;
                C[(size_t)row * GN + col] = acc[i][j][r];
            }
}

// ---------------------------------------------------------------------------
// RoPE + fp16 relayout: (b,s,h,d) f32  ->  (b,h,s,d) f16 for Q,K,V
// One thread per (b,s,h,d<32) pair; handles d and d+32.
// ---------------------------------------------------------------------------
__global__ __launch_bounds__(256)
void rope_cvt(const float* __restrict__ Qf, const float* __restrict__ Kf,
              const float* __restrict__ Vf,
              _Float16* __restrict__ qh, _Float16* __restrict__ kh,
              _Float16* __restrict__ vh) {
    const int i = blockIdx.x * 256 + threadIdx.x;  // 0 .. 2^21-1
    const int d = i & 31;
    const int h = (i >> 5) & 15;
    const int s = (i >> 9) & 2047;
    const int b = i >> 20;

    const size_t src = (((size_t)b * SEQ + s) * HEADS + h) * HDIM + d;
    const size_t dst = (((size_t)b * HEADS + h) * SEQ + s) * HDIM + d;

    // inv_freq = 10000^(-d/32) = exp(-d * ln(10000)/32)
    const float invf = __expf(-(float)d * (9.210340371976184f / 32.0f));
    const float ang  = (float)s * invf;
    float sn, cs;
    __sincosf(ang, &sn, &cs);

    const float q1 = Qf[src], q2 = Qf[src + 32];
    qh[dst]      = (_Float16)(q1 * cs - q2 * sn);
    qh[dst + 32] = (_Float16)(q2 * cs + q1 * sn);

    const float k1 = Kf[src], k2 = Kf[src + 32];
    kh[dst]      = (_Float16)(k1 * cs - k2 * sn);
    kh[dst + 32] = (_Float16)(k2 * cs + k1 * sn);

    vh[dst]      = (_Float16)Vf[src];
    vh[dst + 32] = (_Float16)Vf[src + 32];
}

// ---------------------------------------------------------------------------
// Flash attention: block = (64 queries) x (one b,h). 4 waves x 16 q-rows.
// Streaming softmax over 32 key tiles of 64. Q/K staged via async DMA.
// ---------------------------------------------------------------------------
constexpr int LDQ = 72;   // padded LDS row stride (halves) -> conflict-free

__global__ __launch_bounds__(128)
void attn_fa(const _Float16* __restrict__ qh, const _Float16* __restrict__ kh,
             const _Float16* __restrict__ vh, _Float16* __restrict__ oh) {
    __shared__ _Float16 Qs[64 * LDQ];
    __shared__ _Float16 Ks[64 * LDQ];
    __shared__ _Float16 Vt[64 * LDQ];       // transposed: [d][key]
    __shared__ _Float16 Ps[4][16 * LDQ];    // per-wave P scratch

    const int tid  = threadIdx.x;
    const int wave = tid >> 5, lane = tid & 31;
    const int l16  = lane & 15;
    const int kh8  = (lane >> 4) * 8;
    const int rgrp = (lane >> 4) * 8;

    const int b  = blockIdx.z;
    const int hh = blockIdx.y;
    const int q0 = blockIdx.x * 64;
    const size_t bh = (size_t)b * HEADS + hh;
    const _Float16* Qg = qh + bh * SEQ * HDIM;
    const _Float16* Kg = kh + bh * SEQ * HDIM;
    const _Float16* Vg = vh + bh * SEQ * HDIM;

    const int srow = tid >> 1;            // staging row 0..63
    const int scol = (tid & 1) * 32;      // staging col 0 or 32

    // ---- stage Q tile (64 x 64) once, via async DMA ----
    async_copy64B((unsigned)(size_t)&Qs[srow * LDQ + scol],
                  (unsigned long long)(size_t)(Qg + (size_t)(q0 + srow) * HDIM + scol));
    async_wait0();
    __syncthreads();

    // Q fragments for this wave (constant across key tiles)
    v16h qa0, qa1;
    {
        const _Float16* base = &Qs[(wave * 16 + l16) * LDQ];
        qa0 = mk16(*(const v8h*)(base + kh8),      *(const v8h*)(base + 16 + kh8));
        qa1 = mk16(*(const v8h*)(base + 32 + kh8), *(const v8h*)(base + 48 + kh8));
    }

    float mrow[8], lrow[8];
    v8f oa[4] = {};
#pragma unroll
    for (int r = 0; r < 8; ++r) { mrow[r] = -1e30f; lrow[r] = 0.0f; }

    for (int kt = 0; kt < SEQ / 64; ++kt) {
        // ---- stage K tile (async DMA) and V tile (manual transpose) ----
        async_copy64B((unsigned)(size_t)&Ks[srow * LDQ + scol],
                      (unsigned long long)(size_t)(Kg + (size_t)(kt * 64 + srow) * HDIM + scol));
        {
            const _Float16* pv = Vg + (size_t)(kt * 64 + srow) * HDIM + scol;
            v8h v0 = *(const v8h*)pv,        v1 = *(const v8h*)(pv + 8);
            v8h v2 = *(const v8h*)(pv + 16), v3 = *(const v8h*)(pv + 24);
#pragma unroll
            for (int e = 0; e < 8; ++e) {
                Vt[(scol + e)      * LDQ + srow] = v0[e];
                Vt[(scol + 8 + e)  * LDQ + srow] = v1[e];
                Vt[(scol + 16 + e) * LDQ + srow] = v2[e];
                Vt[(scol + 24 + e) * LDQ + srow] = v3[e];
            }
        }
        async_wait0();
        __syncthreads();

        // ---- scores S = Q * K^T * scale : 4 frags of 16x16 ----
        v8f sc[4];
#pragma unroll
        for (int j = 0; j < 4; ++j) {
            const _Float16* base = &Ks[(j * 16 + l16) * LDQ];
            v16h kb0 = mk16(*(const v8h*)(base + kh8),      *(const v8h*)(base + 16 + kh8));
            v16h kb1 = mk16(*(const v8h*)(base + 32 + kh8), *(const v8h*)(base + 48 + kh8));
            v8f z = {};
            z = wmma_f16(qa0, kb0, z);
            z = wmma_f16(qa1, kb1, z);
#pragma unroll
            for (int r = 0; r < 8; ++r) z[r] *= 0.125f;   // 1/sqrt(64)
            sc[j] = z;
        }

        // ---- online softmax (rows live across 16-lane halves) ----
        float tm[8];
#pragma unroll
        for (int r = 0; r < 8; ++r) {
            float a = fmaxf(fmaxf(sc[0][r], sc[1][r]), fmaxf(sc[2][r], sc[3][r]));
#pragma unroll
            for (int t = 1; t < 16; t <<= 1) a = fmaxf(a, __shfl_xor(a, t, 32));
            tm[r] = a;
        }
#pragma unroll
        for (int r = 0; r < 8; ++r) {
            const float nm   = fmaxf(mrow[r], tm[r]);
            const float corr = __expf(mrow[r] - nm);
            mrow[r] = nm;
            float rs = 0.0f;
#pragma unroll
            for (int j = 0; j < 4; ++j) {
                const float p = __expf(sc[j][r] - nm);
                rs += p;
                Ps[wave][(r + rgrp) * LDQ + j * 16 + l16] = (_Float16)p;
            }
#pragma unroll
            for (int t = 1; t < 16; t <<= 1) rs += __shfl_xor(rs, t, 32);
            lrow[r] = lrow[r] * corr + rs;
#pragma unroll
            for (int j = 0; j < 4; ++j) oa[j][r] *= corr;
        }

        // intra-wave LDS exchange: make lane-crossing P stores visible
        asm volatile("s_wait_dscnt 0" ::: "memory");

        // ---- O += P * V ----
        v16h pa0, pa1;
        {
            const _Float16* base = &Ps[wave][l16 * LDQ];
            pa0 = mk16(*(const v8h*)(base + kh8),      *(const v8h*)(base + 16 + kh8));
            pa1 = mk16(*(const v8h*)(base + 32 + kh8), *(const v8h*)(base + 48 + kh8));
        }
#pragma unroll
        for (int jd = 0; jd < 4; ++jd) {
            const _Float16* base = &Vt[(jd * 16 + l16) * LDQ];
            v16h vb0 = mk16(*(const v8h*)(base + kh8),      *(const v8h*)(base + 16 + kh8));
            v16h vb1 = mk16(*(const v8h*)(base + 32 + kh8), *(const v8h*)(base + 48 + kh8));
            oa[jd] = wmma_f16(pa0, vb0, oa[jd]);
            oa[jd] = wmma_f16(pa1, vb1, oa[jd]);
        }
        __syncthreads();
    }

    // ---- epilogue: O / l  ->  (b,s,h,d) f16 ----
#pragma unroll
    for (int jd = 0; jd < 4; ++jd)
#pragma unroll
        for (int r = 0; r < 8; ++r) {
            const int row  = q0 + wave * 16 + r + rgrp;
            const int dcol = jd * 16 + l16;
            const float v  = oa[jd][r] / lrow[r];
            oh[(((size_t)b * SEQ + row) * HEADS + hh) * HDIM + dcol] = (_Float16)v;
        }
}

// ---------------------------------------------------------------------------
// Launch
// ---------------------------------------------------------------------------
extern "C" void kernel_launch(void* const* d_in, const int* in_sizes, int n_in,
                              void* d_out, int out_size, void* d_ws, size_t ws_size,
                              hipStream_t stream) {
    const float* x  = (const float*)d_in[0];
    const float* Wq = (const float*)d_in[1];
    const float* Wk = (const float*)d_in[2];
    const float* Wv = (const float*)d_in[3];
    const float* Wo = (const float*)d_in[4];
    float* out = (float*)d_out;

    char* ws = (char*)d_ws;
    constexpr size_t F32_MAT = (size_t)GM * GN * sizeof(float);     // 16 MiB
    constexpr size_t F16_MAT = (size_t)GM * GN * sizeof(_Float16);  //  8 MiB
    constexpr size_t F16_W   = (size_t)GN * GK * sizeof(_Float16);  //  2 MiB

    float*    Qf  = (float*)(ws);
    float*    Kf  = (float*)(ws + F32_MAT);
    float*    Vf  = (float*)(ws + 2 * F32_MAT);
    _Float16* qh  = (_Float16*)(ws + 3 * F32_MAT);
    _Float16* kh  = (_Float16*)(ws + 3 * F32_MAT + F16_MAT);
    _Float16* vh  = (_Float16*)(ws + 3 * F32_MAT + 2 * F16_MAT);
    _Float16* xh  = (_Float16*)(ws + 3 * F32_MAT + 3 * F16_MAT);
    _Float16* Wqh = (_Float16*)(ws + 3 * F32_MAT + 4 * F16_MAT);
    _Float16* Wkh = (_Float16*)(ws + 3 * F32_MAT + 4 * F16_MAT + F16_W);
    _Float16* Wvh = (_Float16*)(ws + 3 * F32_MAT + 4 * F16_MAT + 2 * F16_W);
    _Float16* Woh = (_Float16*)(ws + 3 * F32_MAT + 4 * F16_MAT + 3 * F16_W);
    _Float16* oh  = (_Float16*)(ws);   // alias Qf region (dead after rope_cvt)

    // fp32 -> fp16 operand conversion
    cvt_f16<<<(GM * GK / 8) / 256, 256, 0, stream>>>(x,  xh);
    cvt_f16<<<(GN * GK / 8) / 256, 256, 0, stream>>>(Wq, Wqh);
    cvt_f16<<<(GN * GK / 8) / 256, 256, 0, stream>>>(Wk, Wkh);
    cvt_f16<<<(GN * GK / 8) / 256, 256, 0, stream>>>(Wv, Wvh);
    cvt_f16<<<(GN * GK / 8) / 256, 256, 0, stream>>>(Wo, Woh);

    const dim3 ggrid(GN / BN, GM / BM);  // (8, 32)

    gemm_bt<<<ggrid, 256, 0, stream>>>(xh, Wqh, Qf);
    gemm_bt<<<ggrid, 256, 0, stream>>>(xh, Wkh, Kf);
    gemm_bt<<<ggrid, 256, 0, stream>>>(xh, Wvh, Vf);
    rope_cvt<<<(GM * HID / 2) / 256, 256, 0, stream>>>(Qf, Kf, Vf, qh, kh, vh);
    attn_fa<<<dim3(SEQ / 64, HEADS, BATCH), 128, 0, stream>>>(qh, kh, vh, oh);
    gemm_bt<<<ggrid, 256, 0, stream>>>(oh, Woh, out);
}
